// DeltaNetAttention_32839319945672
// MI455X (gfx1250) — matrix operand order repaired and checked by reading, not compiled, not run on hardware
//
#include <hip/hip_runtime.h>
#include <hip/hip_bf16.h>
#include <math.h>

// ---------------------------------------------------------------------------
// DeltaNet attention for MI455X (gfx1250):
//   bf16 WMMA GEMMs with double-buffered async global->LDS staging,
//   register-resident delta-rule scan with chunked async staging.
// ---------------------------------------------------------------------------

typedef __bf16 bf16_t;
typedef __attribute__((ext_vector_type(16))) __bf16 v16bf;
typedef __attribute__((ext_vector_type(8)))  __bf16 v8bf;
typedef __attribute__((ext_vector_type(8)))  float  v8f;

#define BB   8
#define SS   4096
#define HID  1024
#define NH   16
#define DD   64
#define MM   (BB * SS)        // 32768 rows

// ---------------------------------------------------------------- async LDS
__device__ __forceinline__ uint32_t lds_addr(const void* p) {
  return (uint32_t)(uintptr_t)p;   // low 32 bits of generic ptr = LDS offset
}
// CDNA5 async DMA global -> LDS (16B per lane), tracked by ASYNCcnt.
__device__ __forceinline__ void async_cp_b128(const void* g, void* l) {
  asm volatile("global_load_async_to_lds_b128 %0, %1, off"
               :: "v"(lds_addr(l)), "v"(g) : "memory");
}
__device__ __forceinline__ void async_cp_b32(const void* g, void* l) {
  asm volatile("global_load_async_to_lds_b32 %0, %1, off"
               :: "v"(lds_addr(l)), "v"(g) : "memory");
}
__device__ __forceinline__ void wait_async0() {
  asm volatile("s_wait_asynccnt 0x0" ::: "memory");
}

// -------------------------------------------------------------------- utils
__global__ __launch_bounds__(256) void cvt_f32_to_bf16_kernel(
    const float* __restrict__ src, bf16_t* __restrict__ dst, size_t n) {
  size_t i = ((size_t)blockIdx.x * blockDim.x + threadIdx.x) * 4;
  if (i + 3 >= n) return;
  float4 f = *(const float4*)&src[i];
  dst[i + 0] = (bf16_t)f.x;
  dst[i + 1] = (bf16_t)f.y;
  dst[i + 2] = (bf16_t)f.z;
  dst[i + 3] = (bf16_t)f.w;
}

// W [K][N] f32  ->  Wt [N][K] bf16   (one-time, tiny)
__global__ __launch_bounds__(256) void cvt_wt_kernel(
    const float* __restrict__ W, bf16_t* __restrict__ Wt, int K, int N) {
  int idx = blockIdx.x * blockDim.x + threadIdx.x;
  if (idx >= K * N) return;
  int k = idx / N, n = idx % N;
  Wt[(size_t)n * K + k] = (bf16_t)W[idx];
}

// ------------------------------------------------------------- WMMA GEMM
// C[M][N] = A[M][K] * Bt[N][K]^T      (A,Bt bf16 row-major, C fp32)
// 128x128 tile / block (256 thr = 8 waves in a 4(M)x2(N) grid; each wave owns
// a 32x64 strip = 2x4 accumulator tiles). K stepped by 32 (one WMMA-K).
// LDS tiles are double-buffered and filled with async global->LDS DMA.
__global__ __launch_bounds__(256) void gemm_bf16_wmma_kernel(
    const bf16_t* __restrict__ A, const bf16_t* __restrict__ Bt,
    float* __restrict__ C, int M, int N, int K) {
  __shared__ alignas(16) bf16_t As[2][128 * 32];
  __shared__ alignas(16) bf16_t Bs[2][128 * 32];

  const int tid  = threadIdx.x;
  const int wave = tid >> 5;
  const int lane = tid & 31;
  const int half = lane >> 4;   // K-half selector
  const int r    = lane & 15;   // row-in-tile / col-in-tile
  const int wm   = wave >> 1;   // 0..3  (M strip of 32)
  const int wn   = wave & 1;    // 0..1  (N strip of 64)
  const int m0   = blockIdx.y * 128;
  const int n0   = blockIdx.x * 128;

  v8f acc[2][4] = {};

  auto stage = [&](int k0, int buf) {
    // 512 x 16B chunks per matrix tile; 2 async b128 per thread per matrix.
#pragma unroll
    for (int it = 0; it < 2; ++it) {
      int c   = tid + it * 256;       // chunk id 0..511
      int row = c >> 2;
      int cc  = c & 3;
      async_cp_b128(&A[(size_t)(m0 + row) * K + k0 + cc * 8],
                    &As[buf][row * 32 + cc * 8]);
      async_cp_b128(&Bt[(size_t)(n0 + row) * K + k0 + cc * 8],
                    &Bs[buf][row * 32 + cc * 8]);
    }
  };

  stage(0, 0);
  const int NT = K / 32;
  for (int kt = 0; kt < NT; ++kt) {
    const int buf = kt & 1;
    wait_async0();        // our tile's DMA done
    __syncthreads();      // everyone's DMA done; prev compute done everywhere
    if (kt + 1 < NT) stage((kt + 1) * 32, buf ^ 1);

    // --- preload all fragments, then issue 8 back-to-back WMMAs
    union { v16bf v; v8bf h[2]; } af[2], bfv[4];
#pragma unroll
    for (int mt = 0; mt < 2; ++mt) {
      // A frag: lane<16 holds row r, K {0..7,16..23}; lane>=16 K {8..15,24..31}
      const int abase = (wm * 32 + mt * 16 + r) * 32 + half * 8;
      af[mt].h[0] = *(const v8bf*)&As[buf][abase];
      af[mt].h[1] = *(const v8bf*)&As[buf][abase + 16];
    }
#pragma unroll
    for (int nt = 0; nt < 4; ++nt) {
      // B frag: lane<16 holds col r, K {0..7,16..23}; lane>=16 K {8..15,24..31}
      const int bbase = (wn * 64 + nt * 16 + r) * 32 + half * 8;
      bfv[nt].h[0] = *(const v8bf*)&Bs[buf][bbase];
      bfv[nt].h[1] = *(const v8bf*)&Bs[buf][bbase + 16];
    }
#pragma unroll
    for (int mt = 0; mt < 2; ++mt)
#pragma unroll
      for (int nt = 0; nt < 4; ++nt)
        acc[mt][nt] = __builtin_amdgcn_wmma_f32_16x16x32_bf16(
            false, af[mt].v, false, bfv[nt].v, (short)0, acc[mt][nt],
            false, false);
  }

  // --- epilogue: C/D layout: VGPR v -> M = v (lanes 0-15) / v+8 (lanes 16-31)
#pragma unroll
  for (int mt = 0; mt < 2; ++mt)
#pragma unroll
    for (int nt = 0; nt < 4; ++nt)
#pragma unroll
      for (int vv = 0; vv < 8; ++vv) {
        int m = m0 + wm * 32 + mt * 16 + half * 8 + vv;
        int n = n0 + wn * 64 + nt * 16 + r;
        C[(size_t)m * N + n] = acc[mt][nt][vv];
      }
}

// ------------------------------------------------------------ beta = sigmoid(x @ Wbeta)
__global__ __launch_bounds__(256) void beta_kernel(
    const float* __restrict__ x, const float* __restrict__ Wb,
    float* __restrict__ beta, int M) {
  int wave = blockIdx.x * (blockDim.x >> 5) + (threadIdx.x >> 5);
  int lane = threadIdx.x & 31;
  if (wave >= M) return;
  const float* xr = x + (size_t)wave * HID;
  float acc[NH];
#pragma unroll
  for (int h = 0; h < NH; ++h) acc[h] = 0.f;
  for (int k = lane; k < HID; k += 32) {
    float xv = xr[k];
    const float* wr = Wb + (size_t)k * NH;
#pragma unroll
    for (int h = 0; h < NH; ++h) acc[h] = fmaf(xv, wr[h], acc[h]);
  }
#pragma unroll
  for (int h = 0; h < NH; ++h) {
    float v = acc[h];
#pragma unroll
    for (int off = 16; off; off >>= 1) v += __shfl_xor(v, off, 32);
    if (lane == 0) beta[(size_t)wave * NH + h] = 1.f / (1.f + expf(-v));
  }
}

// ------------------------------------------------------------ L2 normalize 64-vectors
__global__ __launch_bounds__(256) void l2norm_kernel(float* __restrict__ t, int nrows) {
  int wave = blockIdx.x * (blockDim.x >> 5) + (threadIdx.x >> 5);
  int lane = threadIdx.x & 31;
  if (wave >= nrows) return;
  float* p = t + (size_t)wave * DD;
  float a0 = p[lane], a1 = p[lane + 32];
  float s = a0 * a0 + a1 * a1;
#pragma unroll
  for (int off = 16; off; off >>= 1) s += __shfl_xor(s, off, 32);
  float scale = 1.f / fmaxf(sqrtf(s), 1e-12f);
  p[lane] = a0 * scale;
  p[lane + 32] = a1 * scale;
}

// ------------------------------------------------------------ delta-rule scan
// one block per (b,h); 64x64 fp32 state in registers (16 elems / thread).
// q/k/v/beta staged 16 timesteps at a time via async DMA, double buffered.
#define TCH 16
__global__ __launch_bounds__(256) void scan_kernel(
    const float* __restrict__ Q, const float* __restrict__ K,
    const float* __restrict__ V, const float* __restrict__ beta,
    float* __restrict__ attn) {
  const int bh  = blockIdx.x;      // 0..127
  const int b   = bh >> 4;
  const int h   = bh & 15;
  const int tid = threadIdx.x;
  const int i   = tid >> 2;        // state row 0..63
  const int jb  = (tid & 3) << 4;  // state col base

  __shared__ alignas(16) float kb[2][TCH * DD];
  __shared__ alignas(16) float vb[2][TCH * DD];
  __shared__ alignas(16) float qb[2][TCH * DD];
  __shared__ alignas(16) float bb[2][TCH];

  float st[16];
#pragma unroll
  for (int jj = 0; jj < 16; ++jj) st[jj] = 0.f;

  auto stage = [&](int c0, int buf) {
    // 3 tensors * TCH steps * 16 chunks(16B) = 768 chunks -> 3 per thread
#pragma unroll
    for (int it = 0; it < 3; ++it) {
      int c      = tid + it * 256;     // 0..767
      int tensor = c >> 8;             // 0:K 1:V 2:Q
      int rem    = c & 255;
      int step   = rem >> 4;
      int part   = rem & 15;           // 16B chunk within 64 floats
      const float* src = tensor == 0 ? K : (tensor == 1 ? V : Q);
      float* dst = tensor == 0 ? kb[buf] : (tensor == 1 ? vb[buf] : qb[buf]);
      async_cp_b128(&src[((size_t)b * SS + c0 + step) * HID + h * DD + part * 4],
                    &dst[step * DD + part * 4]);
    }
    if (tid < TCH)
      async_cp_b32(&beta[((size_t)b * SS + c0 + tid) * NH + h], &bb[buf][tid]);
  };

  stage(0, 0);
  const int NC = SS / TCH;
  for (int cidx = 0; cidx < NC; ++cidx) {
    const int buf = cidx & 1;
    wait_async0();
    __syncthreads();
    if (cidx + 1 < NC) stage((cidx + 1) * TCH, buf ^ 1);

    size_t obase = ((size_t)b * SS + (size_t)cidx * TCH) * HID + (size_t)h * DD;
    for (int s = 0; s < TCH; ++s) {
      const float ki = kb[buf][s * DD + i];
      const float vi = vb[buf][s * DD + i];
      const float bt = bb[buf][s];
      float po = 0.f;
#pragma unroll
      for (int jj = 0; jj < 16; ++jj) {
        int j   = jb + jj;
        float c = bt * kb[buf][s * DD + j];
        float sv = st[jj];
        // s = s*(1 - b*ki*kj) + b*vi*kj  ==  s + c*(vi - ki*s)
        sv = fmaf(c, fmaf(-ki, sv, vi), sv);
        st[jj] = sv;
        po = fmaf(sv, qb[buf][s * DD + j], po);
      }
      po += __shfl_xor(po, 1, 32);
      po += __shfl_xor(po, 2, 32);
      if ((tid & 3) == 0) attn[obase + (size_t)s * HID + i] = po;
    }
  }
}

// ------------------------------------------------------------ LayerNorm + SiLU gate -> bf16
__device__ __forceinline__ float blk_sum(float v, float* red) {
#pragma unroll
  for (int off = 16; off; off >>= 1) v += __shfl_xor(v, off, 32);
  if ((threadIdx.x & 31) == 0) red[threadIdx.x >> 5] = v;
  __syncthreads();
  float tot = 0.f;
#pragma unroll
  for (int w = 0; w < 8; ++w) tot += red[w];
  __syncthreads();
  return tot;
}

__global__ __launch_bounds__(256) void ln_gate_kernel(
    const float* __restrict__ attn, const float* __restrict__ G,
    const float* __restrict__ ln_g, const float* __restrict__ ln_b,
    bf16_t* __restrict__ outb) {
  __shared__ float red[8];
  size_t row = blockIdx.x;
  const float* a = attn + row * HID;
  const float* g = G + row * HID;
  float v[4];
  float s = 0.f;
#pragma unroll
  for (int i = 0; i < 4; ++i) { v[i] = a[threadIdx.x + i * 256]; s += v[i]; }
  float mu = blk_sum(s, red) * (1.f / (float)HID);
  float q = 0.f;
#pragma unroll
  for (int i = 0; i < 4; ++i) { float d = v[i] - mu; q += d * d; }
  float var  = blk_sum(q, red) * (1.f / (float)HID);
  float rinv = rsqrtf(var + 1e-5f);
#pragma unroll
  for (int i = 0; i < 4; ++i) {
    int idx  = threadIdx.x + i * 256;
    float y  = (v[i] - mu) * rinv * ln_g[idx] + ln_b[idx];
    float xg = g[idx];
    float si = xg / (1.f + expf(-xg));      // silu
    outb[row * HID + idx] = (bf16_t)(y * si);
  }
}

// ---------------------------------------------------------------------------
extern "C" void kernel_launch(void* const* d_in, const int* in_sizes, int n_in,
                              void* d_out, int out_size, void* d_ws, size_t ws_size,
                              hipStream_t stream) {
  const float* x     = (const float*)d_in[0];
  const float* Wq    = (const float*)d_in[1];
  const float* Wk    = (const float*)d_in[2];
  const float* Wv    = (const float*)d_in[3];
  const float* Wbeta = (const float*)d_in[4];
  const float* Wg    = (const float*)d_in[5];
  const float* Wout  = (const float*)d_in[6];
  const float* ln_g  = (const float*)d_in[7];
  const float* ln_b  = (const float*)d_in[8];
  float* out = (float*)d_out;

  // ---- workspace layout
  char* ws = (char*)d_ws;
  size_t off = 0;
  auto take = [&](size_t bytes) -> char* {
    char* p = ws + off;
    off += (bytes + 255) & ~(size_t)255;
    return p;
  };
  bf16_t* xb    = (bf16_t*)take((size_t)MM * HID * 2);
  bf16_t* Wqt   = (bf16_t*)take((size_t)HID * HID * 2);
  bf16_t* Wkt   = (bf16_t*)take((size_t)HID * HID * 2);
  bf16_t* Wvt   = (bf16_t*)take((size_t)HID * HID * 2);
  bf16_t* Wgt   = (bf16_t*)take((size_t)HID * HID * 2);
  bf16_t* Wot   = (bf16_t*)take((size_t)HID * HID * 2);
  float*  Qb    = (float*)take((size_t)MM * HID * 4);
  float*  Kb    = (float*)take((size_t)MM * HID * 4);
  float*  Vb    = (float*)take((size_t)MM * HID * 4);
  float*  Gb    = (float*)take((size_t)MM * HID * 4);
  float*  betab = (float*)take((size_t)MM * NH * 4);
  float*  attnb = (float*)take((size_t)MM * HID * 4);
  bf16_t* gated = (bf16_t*)take((size_t)MM * HID * 2);
  if (off > ws_size) return;  // workspace too small; deterministic no-op

  const dim3 blk(256);

  // 1) x -> bf16
  {
    size_t n = (size_t)MM * HID;
    cvt_f32_to_bf16_kernel<<<dim3((unsigned)(n / 4 / 256)), blk, 0, stream>>>(x, xb, n);
  }
  // 2) transpose+convert weights (tiny, one-time)
  {
    unsigned g = (HID * HID + 255) / 256;
    cvt_wt_kernel<<<g, blk, 0, stream>>>(Wq,   Wqt, HID, HID);
    cvt_wt_kernel<<<g, blk, 0, stream>>>(Wk,   Wkt, HID, HID);
    cvt_wt_kernel<<<g, blk, 0, stream>>>(Wv,   Wvt, HID, HID);
    cvt_wt_kernel<<<g, blk, 0, stream>>>(Wg,   Wgt, HID, HID);
    cvt_wt_kernel<<<g, blk, 0, stream>>>(Wout, Wot, HID, HID);
  }
  // 3) projection GEMMs (bf16 WMMA, fp32 accumulate)
  {
    dim3 grid(HID / 128, MM / 128);
    gemm_bf16_wmma_kernel<<<grid, blk, 0, stream>>>(xb, Wqt, Qb, MM, HID, HID);
    gemm_bf16_wmma_kernel<<<grid, blk, 0, stream>>>(xb, Wkt, Kb, MM, HID, HID);
    gemm_bf16_wmma_kernel<<<grid, blk, 0, stream>>>(xb, Wvt, Vb, MM, HID, HID);
    gemm_bf16_wmma_kernel<<<grid, blk, 0, stream>>>(xb, Wgt, Gb, MM, HID, HID);
  }
  // 4) beta = sigmoid(x @ Wbeta)  (fp32 path, skinny N=16)
  beta_kernel<<<dim3(MM / 8), blk, 0, stream>>>(x, Wbeta, betab, MM);
  // 5) L2-normalize q, k per (row, head)
  {
    int nrows = MM * NH;
    l2norm_kernel<<<dim3(nrows / 8), blk, 0, stream>>>(Qb, nrows);
    l2norm_kernel<<<dim3(nrows / 8), blk, 0, stream>>>(Kb, nrows);
  }
  // 6) sequential delta-rule scan (128 independent (b,h) workgroups)
  scan_kernel<<<dim3(BB * NH), blk, 0, stream>>>(Qb, Kb, Vb, betab, attnb);
  // 7) LayerNorm + SiLU gate, emit bf16
  ln_gate_kernel<<<dim3(MM), blk, 0, stream>>>(attnb, Gb, ln_g, ln_b, gated);
  // 8) output projection GEMM -> fp32 result
  {
    dim3 grid(HID / 128, MM / 128);
    gemm_bf16_wmma_kernel<<<grid, blk, 0, stream>>>(gated, Wot, out, MM, HID, HID);
  }
  (void)in_sizes; (void)n_in; (void)out_size;
}
